// Seq2Seq_58437325029425
// MI455X (gfx1250) — matrix operand order repaired and checked by reading, not compiled, not run on hardware
//
#include <hip/hip_runtime.h>
#include <hip/hip_bf16.h>

// Problem constants (from reference setup)
#define BATCH     16384
#define T_STEPS   30
#define IN_DIM    64
#define H_DIM     256
#define G_DIM     768      // 3*H
#define NT_TOTAL  48       // 768 / 16 N-tiles
#define NT_PER_WAVE 6      // 48 / 8 waves
#define ROWS      16       // batch rows per workgroup (one WMMA M tile)

typedef float  v8f   __attribute__((ext_vector_type(8)));
typedef __bf16 v16bf __attribute__((ext_vector_type(16)));

union Frag {
    unsigned u[8];
    uint4    q[2];
    v16bf    bf;
};

// f32 -> bf16 round-to-nearest-even, packed pair into one dword
__device__ __forceinline__ unsigned pack_bf16(float a, float b) {
    unsigned ua = __builtin_bit_cast(unsigned, a);
    unsigned ub = __builtin_bit_cast(unsigned, b);
    ua = (ua + 0x7FFFu + ((ua >> 16) & 1u)) >> 16;
    ub = (ub + 0x7FFFu + ((ub >> 16) & 1u)) >> 16;
    return ua | (ub << 16);
}

__device__ __forceinline__ float sigmoidf_fast(float x) {
    return 1.f / (1.f + __expf(-x));
}
__device__ __forceinline__ float tanhf_fast(float x) {
    return 2.f / (1.f + __expf(-2.f * x)) - 1.f;
}

// Fragment-order index helpers.  A dword d in [0, NKT*256):
//   kt = d>>8, lane = (d>>3)&31, v = d&7
//   row r = lane&15, hi = lane>>4, g = v>>2, j = v&3
//   covers A[r][k0], A[r][k0+1] with k0 = kt*32 + g*16 + hi*8 + j*2
// This matches the 16-bit A 16x32 VGPR layout (ISA 7.12.2).
__device__ __forceinline__ void frag_index(int d, int& r, int& k0) {
    int kt = d >> 8;
    int lane = (d >> 3) & 31;
    int v = d & 7;
    r = lane & 15;
    int hi = lane >> 4;
    int g = v >> 2, j = v & 3;
    k0 = kt * 32 + g * 16 + hi * 8 + j * 2;
}

// -----------------------------------------------------------------------------
// Weight pre-pack: W is (768, K) row-major f32. We need B = W^T (K x 768) in
// bf16, stored in WMMA B-fragment order: per (kt, nt) 32x16 tile, 32 lanes x 8
// dwords contiguous.  Lane layout per ISA 7.12.2 (16-bit B):
//   lanes 0-15  : column n = lane,    K = (v>>2)*16 + 2(v&3), +1   (v = vgpr 0..7)
//   lanes 16-31 : column n = lane-16, K = (v>>2)*16 + 8 + 2(v&3), +1
// -----------------------------------------------------------------------------
__global__ void pack_weights_kernel(const float* __restrict__ W,
                                    unsigned* __restrict__ dst, int K) {
    int d = blockIdx.x * 256 + threadIdx.x;
    int total = (K / 32) * NT_TOTAL * 256;
    if (d >= total) return;
    int block  = d >> 8;
    int within = d & 255;
    int lane = within >> 3;
    int v    = within & 7;
    int kt = block / NT_TOTAL;
    int nt = block % NT_TOTAL;
    int hi = lane >> 4;
    int n  = lane & 15;
    int k0  = kt * 32 + (v >> 2) * 16 + hi * 8 + 2 * (v & 3);
    int col = nt * 16 + n;
    // B[k][col] = W[col][k]
    dst[d] = pack_bf16(W[col * K + k0], W[col * K + k0 + 1]);
}

// -----------------------------------------------------------------------------
// One 16xK @ Kx768 GEMM slice per wave (6 N-tiles of 16), bf16 WMMA, f32 acc.
// sAp: LDS A operand already packed in fragment order (NKT*256 dwords).
// wpk: packed bf16 weights in global.  sG: f32 LDS out, row stride G_DIM.
// -----------------------------------------------------------------------------
template <int NKT>
__device__ __forceinline__ void gemm_tile(const unsigned* __restrict__ sAp,
                                          const unsigned* __restrict__ wpk,
                                          float* __restrict__ sG, int tid) {
    const int wave = tid >> 5;
    const int lane = tid & 31;
    const int r  = lane & 15;
    const int hi = lane >> 4;

    v8f acc[NT_PER_WAVE];
#pragma unroll
    for (int t = 0; t < NT_PER_WAVE; ++t)
        acc[t] = v8f{0.f, 0.f, 0.f, 0.f, 0.f, 0.f, 0.f, 0.f};

#pragma unroll 2
    for (int kt = 0; kt < NKT; ++kt) {
        if (kt + 1 < NKT) {
            __builtin_prefetch(
                wpk + ((((kt + 1) * NT_TOTAL) + wave * NT_PER_WAVE) << 8) +
                    (lane << 3),
                0, 1);
        }
        // A fragment: two b128 LDS loads.
        Frag a;
        const uint4* ap = (const uint4*)sAp + (kt << 6) + (lane << 1);
        a.q[0] = ap[0];
        a.q[1] = ap[1];
#pragma unroll
        for (int t = 0; t < NT_PER_WAVE; ++t) {
            int nt = wave * NT_PER_WAVE + t;
            const uint4* bp =
                (const uint4*)wpk + (((kt * NT_TOTAL) + nt) << 6) + (lane << 1);
            Frag b;
            b.q[0] = bp[0];
            b.q[1] = bp[1];
            acc[t] = __builtin_amdgcn_wmma_f32_16x16x32_bf16(
                false, a.bf, false, b.bf, (short)0, acc[t], false, false);
        }
    }

    // Scatter C (16x16 f32) to LDS: VGPR rr -> M = rr + (hi?8:0), N = r.
#pragma unroll
    for (int t = 0; t < NT_PER_WAVE; ++t) {
        int ncol = (wave * NT_PER_WAVE + t) * 16 + r;
#pragma unroll
        for (int rr = 0; rr < 8; ++rr)
            sG[(hi * 8 + rr) * G_DIM + ncol] = acc[t][rr];
    }
}

// -----------------------------------------------------------------------------
// Full GRU cell for a 16-row tile: gi = A@Wih^T, gh = H@Whh^T, fused
// elementwise gate math updating sHf in place, then cooperative repack of the
// new h into bf16 fragment order (sHp) for the next GEMM that consumes it.
// -----------------------------------------------------------------------------
template <int NKT_A>
__device__ __forceinline__ void gru_cell(const unsigned* __restrict__ sAp,
                                         float* __restrict__ sHf,
                                         unsigned* __restrict__ sHp,
                                         const unsigned* __restrict__ wih,
                                         const unsigned* __restrict__ whh,
                                         const float* __restrict__ bih,
                                         const float* __restrict__ bhh,
                                         float* __restrict__ sGI,
                                         float* __restrict__ sGH, int tid) {
    gemm_tile<NKT_A>(sAp, wih, sGI, tid);
    gemm_tile<H_DIM / 32>(sHp, whh, sGH, tid);
    __syncthreads();

    // Elementwise: thread tid owns column j = tid, all 16 rows.
    const int j = tid;
    const float br  = bih[j] + bhh[j];
    const float bz  = bih[256 + j] + bhh[256 + j];
    const float bin = bih[512 + j];
    const float bhn = bhh[512 + j];
#pragma unroll
    for (int m = 0; m < ROWS; ++m) {
        float gr  = sGI[m * G_DIM + j] + sGH[m * G_DIM + j] + br;
        float gz  = sGI[m * G_DIM + 256 + j] + sGH[m * G_DIM + 256 + j] + bz;
        float gin = sGI[m * G_DIM + 512 + j] + bin;
        float ghn = sGH[m * G_DIM + 512 + j] + bhn;
        float rr = sigmoidf_fast(gr);
        float zz = sigmoidf_fast(gz);
        float nn = tanhf_fast(gin + rr * ghn);
        float hp = sHf[m * H_DIM + j];
        sHf[m * H_DIM + j] = (1.f - zz) * nn + zz * hp;
    }
    __syncthreads();

    // Repack new h into bf16 fragment order (shared across all waves).
#pragma unroll
    for (int i = 0; i < H_DIM / 32; ++i) {
        int d = i * 256 + tid;
        int r, k0;
        frag_index(d, r, k0);
        sHp[d] = pack_bf16(sHf[r * H_DIM + k0], sHf[r * H_DIM + k0 + 1]);
    }
    __syncthreads();
}

// -----------------------------------------------------------------------------
// Persistent seq2seq kernel: each workgroup runs 16 batch rows through the
// whole encoder + decoder, hidden state resident in LDS.
// -----------------------------------------------------------------------------
__launch_bounds__(256)
__global__ void seq2seq_kernel(
    const float* __restrict__ in_data, const float* __restrict__ last_loc,
    const unsigned* __restrict__ w_eih0, const unsigned* __restrict__ w_ehh0,
    const unsigned* __restrict__ w_eih1, const unsigned* __restrict__ w_ehh1,
    const unsigned* __restrict__ w_dih0, const unsigned* __restrict__ w_dhh0,
    const unsigned* __restrict__ w_dih1, const unsigned* __restrict__ w_dhh1,
    const float* __restrict__ e_bih0, const float* __restrict__ e_bhh0,
    const float* __restrict__ e_bih1, const float* __restrict__ e_bhh1,
    const float* __restrict__ d_bih0, const float* __restrict__ d_bhh0,
    const float* __restrict__ d_bih1, const float* __restrict__ d_bhh1,
    const float* __restrict__ embW, const float* __restrict__ embB,
    const float* __restrict__ outW, const float* __restrict__ outB,
    float* __restrict__ out) {
    extern __shared__ float smem[];
    float*    sGI  = smem;                          // 16 x 768
    float*    sGH  = sGI + ROWS * G_DIM;            // 16 x 768
    float*    sH0f = sGH + ROWS * G_DIM;            // 16 x 256 f32
    float*    sH1f = sH0f + ROWS * H_DIM;           // 16 x 256 f32
    unsigned* sH0p = (unsigned*)(sH1f + ROWS * H_DIM);   // 2048 dwords
    unsigned* sH1p = sH0p + 8 * 256;                     // 2048 dwords
    unsigned* sXp  = sH1p + 8 * 256;                     // 2048 dwords
    float*    sEmb = (float*)(sXp + 8 * 256);            // 512 embW + 256 embB
    float*    sXd  = sEmb + 768;                         // 16 x 2

    const int tid = threadIdx.x;
    const int b0  = blockIdx.x * ROWS;

    for (int i = tid; i < ROWS * H_DIM; i += 256) {
        sH0f[i] = 0.f;
        sH1f[i] = 0.f;
    }
    for (int i = tid; i < 8 * 256; i += 256) {
        sH0p[i] = 0u;
        sH1p[i] = 0u;
    }
    // Stage embedding weights/bias for the decoder.
    sEmb[tid] = embW[tid];            // embW is 512 floats; tid covers 0..255
    sEmb[256 + tid] = embW[256 + tid];
    sEmb[512 + tid] = embB[tid];
    __syncthreads();

    // ---------------- Encoder ----------------
    for (int t = 0; t < T_STEPS; ++t) {
        // Pack x_t (16 x 64) straight from global into fragment order.
#pragma unroll
        for (int i = 0; i < IN_DIM / 32; ++i) {
            int d = i * 256 + tid;
            int r, k0;
            frag_index(d, r, k0);
            const float* src =
                in_data + (size_t)(b0 + r) * T_STEPS * IN_DIM + t * IN_DIM + k0;
            sXp[d] = pack_bf16(src[0], src[1]);
        }
        __syncthreads();
        gru_cell<IN_DIM / 32>(sXp, sH0f, sH0p, w_eih0, w_ehh0, e_bih0, e_bhh0,
                              sGI, sGH, tid);
        gru_cell<H_DIM / 32>(sH0p, sH1f, sH1p, w_eih1, w_ehh1, e_bih1, e_bhh1,
                             sGI, sGH, tid);
    }

    // ---------------- Decoder ----------------
    if (tid < 32) {
        int m = tid >> 1, n = tid & 1;
        sXd[m * 2 + n] = last_loc[(size_t)(b0 + m) * 2 + n];
    }
    __syncthreads();

    for (int t = 0; t < T_STEPS; ++t) {
        // Embedding (16x2)@(2x256)+b computed directly in fragment order.
#pragma unroll
        for (int i = 0; i < H_DIM / 32; ++i) {
            int d = i * 256 + tid;
            int r, k0;
            frag_index(d, r, k0);
            float x0 = sXd[r * 2], x1 = sXd[r * 2 + 1];
            float e0 = x0 * sEmb[k0 * 2] + x1 * sEmb[k0 * 2 + 1] +
                       sEmb[512 + k0];
            float e1 = x0 * sEmb[(k0 + 1) * 2] + x1 * sEmb[(k0 + 1) * 2 + 1] +
                       sEmb[512 + k0 + 1];
            sXp[d] = pack_bf16(e0, e1);
        }
        __syncthreads();
        gru_cell<H_DIM / 32>(sXp, sH0f, sH0p, w_dih0, w_dhh0, d_bih0, d_bhh0,
                             sGI, sGH, tid);
        gru_cell<H_DIM / 32>(sH0p, sH1f, sH1p, w_dih1, w_dhh1, d_bih1, d_bhh1,
                             sGI, sGH, tid);

        // Output projection (16x256)@(256x2) + b + x_prev ; 32 lanes.
        if (tid < 32) {
            int m = tid >> 1, n = tid & 1;
            float s = outB[n] + sXd[m * 2 + n];
            const float* wrow = outW + n * H_DIM;
            const float* hrow = sH1f + m * H_DIM;
#pragma unroll 8
            for (int k = 0; k < H_DIM; ++k) s += hrow[k] * wrow[k];
            out[(size_t)(b0 + m) * T_STEPS * 2 + t * 2 + n] = s;
            sXd[m * 2 + n] = s;   // next decoder input
        }
        __syncthreads();
    }
}

// -----------------------------------------------------------------------------
extern "C" void kernel_launch(void* const* d_in, const int* in_sizes, int n_in,
                              void* d_out, int out_size, void* d_ws,
                              size_t ws_size, hipStream_t stream) {
    (void)in_sizes; (void)n_in; (void)out_size; (void)ws_size;

    unsigned* ws = (unsigned*)d_ws;
    const size_t SZ64  = (size_t)IN_DIM * G_DIM / 2;  // dwords, K=64
    const size_t SZ256 = (size_t)H_DIM * G_DIM / 2;   // dwords, K=256

    size_t off = 0;
    unsigned* w_eih0 = ws + off; off += SZ64;
    unsigned* w_ehh0 = ws + off; off += SZ256;
    unsigned* w_eih1 = ws + off; off += SZ256;
    unsigned* w_ehh1 = ws + off; off += SZ256;
    unsigned* w_dih0 = ws + off; off += SZ256;
    unsigned* w_dhh0 = ws + off; off += SZ256;
    unsigned* w_dih1 = ws + off; off += SZ256;
    unsigned* w_dhh1 = ws + off; off += SZ256;

    auto pack = [&](int idx, unsigned* dst, int K) {
        int total = (K / 32) * NT_TOTAL * 256;
        pack_weights_kernel<<<(total + 255) / 256, 256, 0, stream>>>(
            (const float*)d_in[idx], dst, K);
    };
    pack(2,  w_eih0, IN_DIM);   // enc_Wih0 (768,64)
    pack(3,  w_ehh0, H_DIM);    // enc_Whh0 (768,256)
    pack(6,  w_eih1, H_DIM);    // enc_Wih1
    pack(7,  w_ehh1, H_DIM);    // enc_Whh1
    pack(10, w_dih0, H_DIM);    // dec_Wih0
    pack(11, w_dhh0, H_DIM);    // dec_Whh0
    pack(14, w_dih1, H_DIM);    // dec_Wih1
    pack(15, w_dhh1, H_DIM);    // dec_Whh1

    const size_t smemBytes =
        (size_t)(2 * ROWS * G_DIM + 2 * ROWS * H_DIM) * sizeof(float) +
        (size_t)(3 * 8 * 256) * sizeof(unsigned) +
        (size_t)(768 + ROWS * 2) * sizeof(float);

    seq2seq_kernel<<<BATCH / ROWS, 256, smemBytes, stream>>>(
        (const float*)d_in[0], (const float*)d_in[1],
        w_eih0, w_ehh0, w_eih1, w_ehh1, w_dih0, w_dhh0, w_dih1, w_dhh1,
        (const float*)d_in[4],  (const float*)d_in[5],
        (const float*)d_in[8],  (const float*)d_in[9],
        (const float*)d_in[12], (const float*)d_in[13],
        (const float*)d_in[16], (const float*)d_in[17],
        (const float*)d_in[18], (const float*)d_in[19],
        (const float*)d_in[20], (const float*)d_in[21],
        (float*)d_out);
}
